// CompactStreamingDecoder_60017872994832
// MI455X (gfx1250) — compile-verified, matrix operand order checked by reading, not run on hardware
//
#include <hip/hip_runtime.h>
#include <hip/hip_bf16.h>

// ---------------- static dims (from reference) ----------------
#define DB 2
#define DI 1024
#define DA 14
#define DCA 128
#define DCT 384
#define DCS 384
#define DCP 16
#define DW 32
#define DH 4
#define DNB 3
#define DL (DI * DA)   // 14336
#define DNW (DL / DW)  // 448
#define BL (DB * DL)   // 28672 rows of atom activations
#define BI (DB * DI)   // 2048 rows of token activations

typedef __bf16 bf16_t;
typedef bf16_t v16bf __attribute__((ext_vector_type(16)));
typedef bf16_t v8bf __attribute__((ext_vector_type(8)));
typedef float v8f __attribute__((ext_vector_type(8)));

#define SHUF16(lo, hi) \
  __builtin_shufflevector((lo), (hi), 0, 1, 2, 3, 4, 5, 6, 7, 8, 9, 10, 11, 12, 13, 14, 15)

__device__ __forceinline__ float wave_sum32(float v) {
#pragma unroll
  for (int m = 16; m >= 1; m >>= 1) v += __shfl_xor(v, m, 32);
  return v;
}

__device__ __forceinline__ float sigmoidf_(float x) { return 1.f / (1.f + __expf(-x)); }

// ---- weight prep: W[K,N] f32 -> WT[N,K] bf16 (once per GEMM, not per M-tile) ----
__global__ __launch_bounds__(256) void k_wt_bf16(const float* __restrict__ W,
                                                 bf16_t* __restrict__ WT, int K, int N) {
  int idx = blockIdx.x * 256 + threadIdx.x;
  if (idx >= K * N) return;
  int k = idx / N;
  int n = idx - k * N;
  WT[(size_t)n * K + k] = (bf16_t)W[idx];
}

// ---------------- WMMA GEMM: C[M,N] f32 = A[M,K] bf16 * WT[N,K] bf16 ----------
// 256 threads = 8 waves; 128x128 tile per workgroup; K stepped by 32.
// Wave (wm,wn) in 4x2 computes a 32x64 sub-tile: 2 A-frags x 4 B-frags = 8 WMMA.
__global__ __launch_bounds__(256) void k_gemm_bb(const bf16_t* __restrict__ A,
                                                 const bf16_t* __restrict__ WT,
                                                 float* __restrict__ C, int M, int N,
                                                 int K) {
  __shared__ bf16_t At[128 * 32];  // [m_local][k_local]
  __shared__ bf16_t Bt[128 * 32];  // [n_local][k_local]
  const int tid = threadIdx.x;
  const int w = tid >> 5;
  const int wm = w >> 1;   // 0..3 -> 32-row band
  const int wn = w & 1;    // 0..1 -> 64-col band
  const int lane = tid & 31;
  const int lrow = lane & 15;
  const int khalf = lane >> 4;
  const int m0 = blockIdx.y * 128;
  const int n0 = blockIdx.x * 128;

  const v8f zero = {0.f, 0.f, 0.f, 0.f, 0.f, 0.f, 0.f, 0.f};
  v8f acc[2][4];
#pragma unroll
  for (int i = 0; i < 2; ++i)
#pragma unroll
    for (int t = 0; t < 4; ++t) acc[i][t] = zero;

  for (int k0 = 0; k0 < K; k0 += 32) {
// A tile: 128x32 bf16 = 512 v8bf chunks; B tile likewise from pre-transposed WT
#pragma unroll
    for (int it = 0; it < 2; ++it) {
      int c = tid + it * 256;
      int r = c >> 2, c8 = (c & 3) * 8;
      *(v8bf*)&At[r * 32 + c8] = *(const v8bf*)&A[(size_t)(m0 + r) * K + k0 + c8];
      *(v8bf*)&Bt[r * 32 + c8] = *(const v8bf*)&WT[(size_t)(n0 + r) * K + k0 + c8];
    }
    __syncthreads();

    // A fragments: lanes 0-15 hold M rows; VGPR0-3 = K(0..15|half), VGPR4-7 = K(16..31)
    const v8bf* ar0 = (const v8bf*)&At[(wm * 32 + lrow) * 32];
    const v8bf* ar1 = (const v8bf*)&At[(wm * 32 + 16 + lrow) * 32];
    v16bf a0 = SHUF16(ar0[khalf], ar0[2 + khalf]);
    v16bf a1 = SHUF16(ar1[khalf], ar1[2 + khalf]);
#pragma unroll
    for (int t = 0; t < 4; ++t) {
      const v8bf* br = (const v8bf*)&Bt[(wn * 64 + t * 16 + lrow) * 32];
      v16bf bfv = SHUF16(br[khalf], br[2 + khalf]);
      acc[0][t] = __builtin_amdgcn_wmma_f32_16x16x32_bf16(false, a0, false, bfv, (short)0,
                                                          acc[0][t], false, false);
      acc[1][t] = __builtin_amdgcn_wmma_f32_16x16x32_bf16(false, a1, false, bfv, (short)0,
                                                          acc[1][t], false, false);
    }
    __syncthreads();
  }

// D layout: VGPR r holds row (khalf*8 + r), column = lrow within each 16x16 tile
#pragma unroll
  for (int i = 0; i < 2; ++i) {
#pragma unroll
    for (int t = 0; t < 4; ++t) {
#pragma unroll
      for (int r = 0; r < 8; ++r) {
        int m = m0 + wm * 32 + i * 16 + khalf * 8 + r;
        int n = n0 + wn * 64 + t * 16 + lrow;
        C[(size_t)m * N + n] = acc[i][t][r];
      }
    }
  }
}

// ---------------- LayerNorm / RMSNorm (wave32 per row) ----------------
__global__ __launch_bounds__(256) void k_ln128_na(const float* __restrict__ X,
                                                  float* __restrict__ Y, int rows) {
  int gw = (blockIdx.x * 256 + threadIdx.x) >> 5;
  int lane = threadIdx.x & 31;
  if (gw >= rows) return;
  const float* x = X + (size_t)gw * DCA;
  float v[4], s = 0.f;
#pragma unroll
  for (int i = 0; i < 4; ++i) {
    v[i] = x[lane + 32 * i];
    s += v[i];
  }
  float mean = wave_sum32(s) * (1.f / 128.f);
  float q = 0.f;
#pragma unroll
  for (int i = 0; i < 4; ++i) {
    float d = v[i] - mean;
    q += d * d;
  }
  float inv = rsqrtf(wave_sum32(q) * (1.f / 128.f) + 1e-5f);
  float* y = Y + (size_t)gw * DCA;
#pragma unroll
  for (int i = 0; i < 4; ++i) y[lane + 32 * i] = (v[i] - mean) * inv;
}

__global__ __launch_bounds__(256) void k_ln128_g_bf16(const float* __restrict__ X,
                                                      const float* __restrict__ g,
                                                      bf16_t* __restrict__ Y, int rows) {
  int gw = (blockIdx.x * 256 + threadIdx.x) >> 5;
  int lane = threadIdx.x & 31;
  if (gw >= rows) return;
  const float* x = X + (size_t)gw * DCA;
  float v[4], s = 0.f;
#pragma unroll
  for (int i = 0; i < 4; ++i) {
    v[i] = x[lane + 32 * i];
    s += v[i];
  }
  float mean = wave_sum32(s) * (1.f / 128.f);
  float q = 0.f;
#pragma unroll
  for (int i = 0; i < 4; ++i) {
    float d = v[i] - mean;
    q += d * d;
  }
  float inv = rsqrtf(wave_sum32(q) * (1.f / 128.f) + 1e-5f);
  bf16_t* y = Y + (size_t)gw * DCA;
#pragma unroll
  for (int i = 0; i < 4; ++i)
    y[lane + 32 * i] = (bf16_t)((v[i] - mean) * inv * g[lane + 32 * i]);
}

__global__ __launch_bounds__(256) void k_rms384_bf16(const float* __restrict__ X,
                                                     const float* __restrict__ g,
                                                     bf16_t* __restrict__ Y, int rows) {
  int gw = (blockIdx.x * 256 + threadIdx.x) >> 5;
  int lane = threadIdx.x & 31;
  if (gw >= rows) return;
  const float* x = X + (size_t)gw * DCT;
  float v[12], s = 0.f;
#pragma unroll
  for (int i = 0; i < 12; ++i) {
    v[i] = x[lane + 32 * i];
    s += v[i] * v[i];
  }
  float inv = rsqrtf(wave_sum32(s) * (1.f / 384.f) + 1e-6f);
  bf16_t* y = Y + (size_t)gw * DCT;
#pragma unroll
  for (int i = 0; i < 12; ++i)
    y[lane + 32 * i] = (bf16_t)(v[i] * inv * g[lane + 32 * i]);
}

// ---------------- elementwise ----------------
__global__ __launch_bounds__(256) void k_copy_f32(float* __restrict__ dst,
                                                  const float* __restrict__ src, int n) {
  int idx = blockIdx.x * 256 + threadIdx.x;
  if (idx < n) dst[idx] = src[idx];
}

__global__ __launch_bounds__(256) void k_cvt_bf16(bf16_t* __restrict__ dst,
                                                  const float* __restrict__ src, int n) {
  int idx = blockIdx.x * 256 + threadIdx.x;
  if (idx < n) dst[idx] = (bf16_t)src[idx];
}

// a = sigmoid(G + bg[c]) * an + Bt  -> bf16
__global__ __launch_bounds__(256) void k_ada_combine(const float* __restrict__ G,
                                                     const float* __restrict__ bg,
                                                     const float* __restrict__ an,
                                                     const float* __restrict__ Bt,
                                                     bf16_t* __restrict__ outb, int n) {
  int idx = blockIdx.x * 256 + threadIdx.x;
  if (idx >= n) return;
  int c = idx & (DCA - 1);
  outb[idx] = (bf16_t)(sigmoidf_(G[idx] + bg[c]) * an[idx] + Bt[idx]);
}

// Q[b,l,c] += T[b, l/14, c]
__global__ __launch_bounds__(256) void k_upcast_add(float* __restrict__ Q,
                                                    const float* __restrict__ T, int n) {
  int idx = blockIdx.x * 256 + threadIdx.x;
  if (idx >= n) return;
  int c = idx & (DCA - 1);
  int row = idx >> 7;
  int b = row / DL;
  int tok = (row - b * DL) / DA;
  Q[idx] += T[((size_t)(b * DI + tok)) * DCA + c];
}

// Q += sigmoid(gpre + gb[c]) * val
__global__ __launch_bounds__(256) void k_gated_add(float* __restrict__ Q,
                                                   const float* __restrict__ gpre,
                                                   const float* __restrict__ gb,
                                                   const float* __restrict__ val, int n) {
  int idx = blockIdx.x * 256 + threadIdx.x;
  if (idx >= n) return;
  int c = idx & (DCA - 1);
  Q[idx] += sigmoidf_(gpre[idx] + gb[c]) * val[idx];
}

// h = silu(u1) * u2 -> bf16
__global__ __launch_bounds__(256) void k_swiglu(const float* __restrict__ u1,
                                                const float* __restrict__ u2,
                                                bf16_t* __restrict__ h, int n) {
  int idx = blockIdx.x * 256 + threadIdx.x;
  if (idx >= n) return;
  float u = u1[idx];
  h[idx] = (bf16_t)((u * sigmoidf_(u)) * u2[idx]);
}

// bias[n,h,q,k] = sum_c LN16(P[n,q,k,:]) * Wp[c,h]
__global__ __launch_bounds__(256) void k_pair_bias(const float* __restrict__ P,
                                                   const float* __restrict__ g,
                                                   const float* __restrict__ bb,
                                                   const float* __restrict__ Wp,
                                                   float* __restrict__ bias) {
  int idx = blockIdx.x * 256 + threadIdx.x;  // over 448*32*32
  if (idx >= DNW * DW * DW) return;
  int kk = idx & 31;
  int qq = (idx >> 5) & 31;
  int n = idx >> 10;
  const float* p = P + (size_t)idx * DCP;
  float x[DCP], s = 0.f;
#pragma unroll
  for (int c = 0; c < DCP; ++c) {
    x[c] = p[c];
    s += x[c];
  }
  float mean = s * (1.f / DCP);
  float var = 0.f;
#pragma unroll
  for (int c = 0; c < DCP; ++c) {
    float d = x[c] - mean;
    var += d * d;
  }
  float inv = rsqrtf(var * (1.f / DCP) + 1e-5f);
#pragma unroll
  for (int h = 0; h < DH; ++h) {
    float acc = 0.f;
#pragma unroll
    for (int c = 0; c < DCP; ++c)
      acc += ((x[c] - mean) * inv * g[c] + bb[c]) * Wp[c * DH + h];
    bias[(((size_t)n * DH + h) * DW + qq) * DW + kk] = acc;
  }
}

// per-(window,head) attention core: 128 threads = 4 waves = 4 heads
__global__ __launch_bounds__(128) void k_attn(const float* __restrict__ q,
                                              const float* __restrict__ k,
                                              const float* __restrict__ v,
                                              const float* __restrict__ bias,
                                              bf16_t* __restrict__ obf) {
  __shared__ float ks[DH * DW * DW];
  __shared__ float vs[DH * DW * DW];
  const int h = threadIdx.x >> 5;
  const int lane = threadIdx.x & 31;
  const int n = blockIdx.x % DNW;
  const int b = blockIdx.x / DNW;
  const size_t src = ((size_t)(b * DL + n * DW + lane)) * DCA + h * DW;

  float qreg[DW];
  {
    const float4* qp = (const float4*)(q + src);
    const float4* kp = (const float4*)(k + src);
    const float4* vp = (const float4*)(v + src);
    float4* kd = (float4*)&ks[(h * DW + lane) * DW];
    float4* vd = (float4*)&vs[(h * DW + lane) * DW];
#pragma unroll
    for (int j = 0; j < 8; ++j) {
      float4 qv = qp[j];
      qreg[4 * j + 0] = qv.x;
      qreg[4 * j + 1] = qv.y;
      qreg[4 * j + 2] = qv.z;
      qreg[4 * j + 3] = qv.w;
      kd[j] = kp[j];
      vd[j] = vp[j];
    }
  }
  __syncthreads();

  const float* bp = bias + (((size_t)n * DH + h) * DW + lane) * DW;
  float lg[DW];
  float mx = -1e30f;
#pragma unroll
  for (int kk = 0; kk < DW; ++kk) {
    const float* kr = &ks[(h * DW + kk) * DW];
    float s = 0.f;
#pragma unroll
    for (int d = 0; d < DW; ++d) s += qreg[d] * kr[d];
    s = s * 0.17677669529663687f + bp[kk];  // 1/sqrt(32)
    lg[kk] = s;
    mx = fmaxf(mx, s);
  }
  float den = 0.f;
#pragma unroll
  for (int kk = 0; kk < DW; ++kk) {
    lg[kk] = __expf(lg[kk] - mx);
    den += lg[kk];
  }
  const float idn = 1.f / den;
  float o[DW];
#pragma unroll
  for (int d = 0; d < DW; ++d) o[d] = 0.f;
#pragma unroll
  for (int kk = 0; kk < DW; ++kk) {
    const float p = lg[kk] * idn;
    const float* vr = &vs[(h * DW + kk) * DW];
#pragma unroll
    for (int d = 0; d < DW; ++d) o[d] += p * vr[d];
  }
  bf16_t* op = obf + src;
#pragma unroll
  for (int d = 0; d < DW; ++d) op[d] = (bf16_t)o[d];
}

// A_new = A_I + mean_a(D rows) + S2
__global__ __launch_bounds__(256) void k_final(const float* __restrict__ A_I,
                                               const float* __restrict__ D,
                                               const float* __restrict__ S2,
                                               float* __restrict__ out, int n) {
  int idx = blockIdx.x * 256 + threadIdx.x;
  if (idx >= n) return;
  int c = idx % DCT;
  int bi = idx / DCT;  // b*I + i ; D row base = bi*14 (since L = I*14)
  const float* drow = D + ((size_t)bi * DA) * DCT + c;
  float s = 0.f;
#pragma unroll
  for (int a = 0; a < DA; ++a) s += drow[(size_t)a * DCT];
  out[idx] = A_I[idx] + s * (1.f / (float)DA) + S2[idx];
}

// ---------------- launcher ----------------
extern "C" void kernel_launch(void* const* d_in, const int* in_sizes, int n_in,
                              void* d_out, int out_size, void* d_ws, size_t ws_size,
                              hipStream_t stream) {
  (void)in_sizes;
  (void)n_in;
  (void)out_size;
  (void)ws_size;
  // inputs (setup_inputs order)
  const float* A_I = (const float*)d_in[0];
  const float* S_I = (const float*)d_in[1];
  const float* Q_in = (const float*)d_in[3];
  const float* C_L = (const float*)d_in[4];
  const float* P_LL = (const float*)d_in[5];
  // params: jax pytree sorted-key flatten order
  const float* aWga = (const float*)d_in[8];
  const float* aWk = (const float*)d_in[9];
  const float* aWo = (const float*)d_in[10];
  const float* aWp = (const float*)d_in[11];
  const float* aWq = (const float*)d_in[12];
  const float* aWv = (const float*)d_in[13];
  const float* aaWb = (const float*)d_in[14];
  const float* aaWg = (const float*)d_in[15];
  const float* aabg = (const float*)d_in[16];
  const float* aags = (const float*)d_in[17];
  const float* abga = (const float*)d_in[18];
  const float* alnb = (const float*)d_in[19];
  const float* alng = (const float*)d_in[20];
  const float* tW1 = (const float*)d_in[21];
  const float* tW2 = (const float*)d_in[22];
  const float* tW3 = (const float*)d_in[23];
  const float* tWog = (const float*)d_in[24];
  const float* taWb = (const float*)d_in[25];
  const float* taWg = (const float*)d_in[26];
  const float* tabg = (const float*)d_in[27];
  const float* tags = (const float*)d_in[28];
  const float* tbog = (const float*)d_in[29];
  const float* upW = (const float*)d_in[30];
  const float* upg = (const float*)d_in[31];
  const float* dWd = (const float*)d_in[32];
  const float* dWs = (const float*)d_in[33];
  const float* dsg = (const float*)d_in[34];
  float* out = (float*)d_out;
  char* ws = (char*)d_ws;

  // workspace arena
  constexpr size_t SQB = (size_t)BL * DCA * sizeof(float);  // 14.68 MB
  constexpr size_t SQHB = SQB / 2;
  constexpr size_t S256B = (size_t)BL * 256 * sizeof(float);  // 29.36 MB
  constexpr size_t OFF_Q = 0;
  constexpr size_t OFF_AN = OFF_Q + SQB;
  constexpr size_t OFF_SNB = OFF_AN + SQB;
  constexpr size_t OFF_ABF = OFF_SNB + SQHB;
  constexpr size_t OFF_CB = OFF_ABF + SQHB;
  constexpr size_t OFF_OBF = OFF_CB + SQHB;
  constexpr size_t OFF_F1 = OFF_OBF + SQHB;  // also u1 / gg / D (with F2)
  constexpr size_t OFF_F2 = OFF_F1 + S256B;  // also u2 / go
  constexpr size_t OFF_QB = OFF_F2 + S256B;
  constexpr size_t OFF_KB = OFF_QB + SQB;
  constexpr size_t OFF_VB = OFF_KB + SQB;
  constexpr size_t OFF_BIAS = OFF_VB + SQB;                 // 7.34 MB
  constexpr size_t OFF_HBF = OFF_BIAS + SQHB;               // BL*256 bf16
  constexpr size_t OFF_T = OFF_HBF + (size_t)BL * 256 * 2;  // BI*128 f32
  constexpr size_t OFF_RA = OFF_T + (size_t)BI * DCA * 4;   // BI*384 bf16
  constexpr size_t OFF_RS = OFF_RA + (size_t)BI * DCT * 2;  // BI*384 bf16
  constexpr size_t OFF_S2 = OFF_RS + (size_t)BI * DCS * 2;  // BI*384 f32
  constexpr size_t OFF_WT = OFF_S2 + (size_t)BI * DCT * 4;  // 384*384 bf16 scratch

  float* Qf = (float*)(ws + OFF_Q);
  float* ANf = (float*)(ws + OFF_AN);
  bf16_t* SNB = (bf16_t*)(ws + OFF_SNB);
  bf16_t* ABF = (bf16_t*)(ws + OFF_ABF);
  bf16_t* CB = (bf16_t*)(ws + OFF_CB);
  bf16_t* OBF = (bf16_t*)(ws + OFF_OBF);
  float* F1 = (float*)(ws + OFF_F1);
  float* F2 = (float*)(ws + OFF_F2);
  float* QB = (float*)(ws + OFF_QB);
  float* KB = (float*)(ws + OFF_KB);
  float* VB = (float*)(ws + OFF_VB);
  float* BIAS = (float*)(ws + OFF_BIAS);
  bf16_t* HBF = (bf16_t*)(ws + OFF_HBF);
  float* Tf = (float*)(ws + OFF_T);
  bf16_t* RA = (bf16_t*)(ws + OFF_RA);
  bf16_t* RS = (bf16_t*)(ws + OFF_RS);
  float* S2 = (float*)(ws + OFF_S2);
  bf16_t* WTB = (bf16_t*)(ws + OFF_WT);
  float* Dbuf = F1;  // BL*384 f32 fits in F1+F2 region (used after blocks)

  // convert+transpose weight once, then 128x128-tile WMMA GEMM
  auto gemm = [&](const bf16_t* Ab, const float* Wp_, float* Co, int M, int N, int K) {
    k_wt_bf16<<<(K * N + 255) / 256, 256, 0, stream>>>(Wp_, WTB, K, N);
    dim3 grid(N / 128, M / 128);
    k_gemm_bb<<<grid, dim3(256), 0, stream>>>(Ab, WTB, Co, M, N, K);
  };
  const int EW_QL = (BL * DCA + 255) / 256;

  // init: Q canonical copy, bf16 of C
  k_copy_f32<<<EW_QL, 256, 0, stream>>>(Qf, Q_in, BL * DCA);
  k_cvt_bf16<<<EW_QL, 256, 0, stream>>>(CB, C_L, BL * DCA);

  for (int i = 0; i < DNB; ++i) {
    const size_t oM = (size_t)i * DCA * DCA;  // 128x128 mats
    const size_t oV = (size_t)i * DCA;        // 128 vecs
    // ---- upcast: Q += broadcast( rms(A_I)*up_g @ up_W ) ----
    k_rms384_bf16<<<(BI + 7) / 8, 256, 0, stream>>>(A_I, upg + (size_t)i * DCT, RA, BI);
    gemm(RA, upW + (size_t)i * DCT * DCA, Tf, BI, DCA, DCT);
    k_upcast_add<<<EW_QL, 256, 0, stream>>>(Qf, Tf, BL * DCA);

    // ---- attention: AdaLN ----
    k_ln128_na<<<(BL + 7) / 8, 256, 0, stream>>>(Qf, ANf, BL);
    k_ln128_g_bf16<<<(BL + 7) / 8, 256, 0, stream>>>(C_L, aags + oV, SNB, BL);
    gemm(SNB, aaWg + oM, F1, BL, DCA, DCA);
    gemm(SNB, aaWb + oM, F2, BL, DCA, DCA);
    k_ada_combine<<<EW_QL, 256, 0, stream>>>(F1, aabg + oV, ANf, F2, ABF, BL * DCA);
    // q,k,v projections
    gemm(ABF, aWq + oM, QB, BL, DCA, DCA);
    gemm(ABF, aWk + oM, KB, BL, DCA, DCA);
    gemm(ABF, aWv + oM, VB, BL, DCA, DCA);
    // pair bias + windowed attention core
    k_pair_bias<<<(DNW * DW * DW + 255) / 256, 256, 0, stream>>>(
        P_LL, alng + (size_t)i * DCP, alnb + (size_t)i * DCP, aWp + (size_t)i * DCP * DH,
        BIAS);
    k_attn<<<DB * DNW, 128, 0, stream>>>(QB, KB, VB, BIAS, OBF);
    // gated output: Q += sigmoid(C@Wga + bga) * (o @ Wo)
    gemm(OBF, aWo + oM, F2, BL, DCA, DCA);
    gemm(CB, aWga + oM, F1, BL, DCA, DCA);
    k_gated_add<<<EW_QL, 256, 0, stream>>>(Qf, F1, abga + oV, F2, BL * DCA);

    // ---- conditioned transition ----
    k_ln128_na<<<(BL + 7) / 8, 256, 0, stream>>>(Qf, ANf, BL);
    k_ln128_g_bf16<<<(BL + 7) / 8, 256, 0, stream>>>(C_L, tags + oV, SNB, BL);
    gemm(SNB, taWg + oM, F1, BL, DCA, DCA);
    gemm(SNB, taWb + oM, F2, BL, DCA, DCA);
    k_ada_combine<<<EW_QL, 256, 0, stream>>>(F1, tabg + oV, ANf, F2, ABF, BL * DCA);
    gemm(ABF, tW1 + (size_t)i * DCA * 256, F1, BL, 256, DCA);
    gemm(ABF, tW2 + (size_t)i * DCA * 256, F2, BL, 256, DCA);
    k_swiglu<<<(BL * 256 + 255) / 256, 256, 0, stream>>>(F1, F2, HBF, BL * 256);
    gemm(HBF, tW3 + (size_t)i * 256 * DCA, F1, BL, DCA, 256);
    gemm(CB, tWog + oM, F2, BL, DCA, DCA);
    k_gated_add<<<EW_QL, 256, 0, stream>>>(Qf, F2, tbog + oV, F1, BL * DCA);
  }

  // ---- downcast + output ----
  bf16_t* QBF = ABF;  // reuse as bf16 copy of final Q
  k_cvt_bf16<<<EW_QL, 256, 0, stream>>>(QBF, Qf, BL * DCA);
  gemm(QBF, dWd, Dbuf, BL, DCT, DCA);  // [BL,384] fits in F1+F2
  k_rms384_bf16<<<(BI + 7) / 8, 256, 0, stream>>>(S_I, dsg, RS, BI);
  gemm(RS, dWs, S2, BI, DCT, DCS);
  k_final<<<(BI * DCT + 255) / 256, 256, 0, stream>>>(A_I, Dbuf, S2, out, BI * DCT);
  // second output: final Q_L
  k_copy_f32<<<EW_QL, 256, 0, stream>>>(out + (size_t)BI * DCT, Qf, BL * DCA);
}
